// Multi_Head_Attention_43130061587005
// MI455X (gfx1250) — matrix operand order repaired and checked by reading, not compiled, not run on hardware
//
#include <hip/hip_runtime.h>

// ---------------------------------------------------------------------------
// Causal multi-head attention forward on MI455X (gfx1250, wave32, WMMA).
//   1) one-shot fp32->bf16 weight conversion (WQ/WK/WV/WO)
//   2) fused head projections as [B*S,1024]x[1024,1024] bf16 GEMM, 64x64/wave
//   3) flash attention: 32 query rows/wave, 32 keys/iter, online softmax,
//      bf16 WMMA + fp32 accum, double-buffered LDS P-staging, K/V prefetch
//   4) output projection, 64x64 tiles/wave, fp32 result
// ---------------------------------------------------------------------------

typedef __bf16 bf16_t;
typedef __attribute__((ext_vector_type(16))) __bf16 v16bf;
typedef __attribute__((ext_vector_type(8)))  __bf16 v8bf;
typedef __attribute__((ext_vector_type(8)))  float  v8f;

#define B_  4
#define S_  2048
#define D_  1024
#define H_  16
#define DK_ 64
#define DV_ 64
#define HD_ 1024   // H_*DV_

static __device__ __forceinline__ v8f wmma_bf16(v16bf a, v16bf b, v8f c) {
  return __builtin_amdgcn_wmma_f32_16x16x32_bf16(false, a, false, b, (short)0, c,
                                                 false, false);
}

// A-operand (16x32 bf16) for lane (m = lane&15, g = lane>>4):
//   a[0..7]  = row[kBase + 16g + i],  a[8..15] = row[kBase + 16g + 8 + i]
static __device__ __forceinline__ v16bf a_tile_f32(const float* row, int kBase, int g) {
  v16bf r;
#pragma unroll
  for (int i = 0; i < 8; ++i) {
    r[i]     = (__bf16)row[kBase + 16 * g + i];
    r[8 + i] = (__bf16)row[kBase + 16 * g + 8 + i];
  }
  return r;
}

static __device__ __forceinline__ v16bf a_tile_bf16(const bf16_t* row, int kBase, int g) {
  v8bf lo = *(const v8bf*)(row + kBase + 16 * g);
  v8bf hi = *(const v8bf*)(row + kBase + 16 * g + 8);
  v16bf r;
#pragma unroll
  for (int i = 0; i < 8; ++i) { r[i] = lo[i]; r[8 + i] = hi[i]; }
  return r;
}

// ---------------------------------------------------------------------------
// Kernel 0: fp32 -> bf16 elementwise (weights), 8 elems/thread
// ---------------------------------------------------------------------------
__global__ __launch_bounds__(256)
void cvt_kernel(const float* __restrict__ src, bf16_t* __restrict__ dst) {
  const size_t i = ((size_t)blockIdx.x * blockDim.x + threadIdx.x) * 8;
  v8bf o;
#pragma unroll
  for (int j = 0; j < 8; ++j) o[j] = (__bf16)src[i + j];
  *(v8bf*)(dst + i) = o;
}

// ---------------------------------------------------------------------------
// Kernel 1: fused projection GEMM  X[B*S,1024] x Wbf[1024,1024]^T
// 64x64 output tile per wave (4 M-tiles x 4 N-tiles, 16 WMMA / k-step).
// Global output column ng = h*64 + f scatters into head-major layouts:
//   mode 0: Out[b,h,s,f]   (Qh / Kh)
//   mode 1: Out[b,h,f,s]   (VhT, transposed for the PV B-operand)
// ---------------------------------------------------------------------------
__global__ __launch_bounds__(32)
void proj_kernel(const float* __restrict__ X, const bf16_t* __restrict__ Wbf,
                 bf16_t* __restrict__ Out, int mode, float outScale) {
  const int lane = threadIdx.x;
  const int m = lane & 15, g = lane >> 4, n = m;
  const size_t rb = (size_t)blockIdx.x * 64;      // flattened B*S row base
  const int nb = blockIdx.y * 64;                 // output column base

  v8f acc[4][4];
#pragma unroll
  for (int mt = 0; mt < 4; ++mt)
#pragma unroll
    for (int nt = 0; nt < 4; ++nt) acc[mt][nt] = (v8f){0};

  for (int kb = 0; kb < D_; kb += 32) {
    v16bf a[4];
#pragma unroll
    for (int mt = 0; mt < 4; ++mt)
      a[mt] = a_tile_f32(X + (rb + mt * 16 + m) * D_, kb, g);
#pragma unroll
    for (int nt = 0; nt < 4; ++nt) {
      v16bf b = *(const v16bf*)(Wbf + (size_t)(nb + nt * 16 + n) * D_ + kb + 16 * g);
#pragma unroll
      for (int mt = 0; mt < 4; ++mt) acc[mt][nt] = wmma_bf16(a[mt], b, acc[mt][nt]);
    }
  }

#pragma unroll
  for (int mt = 0; mt < 4; ++mt)
#pragma unroll
    for (int nt = 0; nt < 4; ++nt) {
      const int ng = nb + nt * 16 + n;
      const int h = ng >> 6, f = ng & 63;
#pragma unroll
      for (int r = 0; r < 8; ++r) {
        const size_t row = rb + mt * 16 + r + 8 * g;
        const size_t bb = row >> 11;              // row / S_
        const size_t s  = row & (S_ - 1);
        const bf16_t v = (__bf16)(acc[mt][nt][r] * outScale);
        if (mode == 0)
          Out[((bb * H_ + h) * S_ + s) * 64 + f] = v;
        else
          Out[((bb * H_ + h) * 64 + f) * S_ + s] = v;
      }
    }
}

// ---------------------------------------------------------------------------
// Kernel 2: flash attention, causal. 32 query rows per wave (2 M-tiles),
// 32 keys per iteration, exp2-domain online softmax (log2e folded into Qh).
// K/V B-operands shared by both M-tiles; double-buffered LDS P-staging;
// next iteration's K/V pulled in early via global_prefetch.
// ---------------------------------------------------------------------------
__global__ __launch_bounds__(32)
void attn_kernel(const bf16_t* __restrict__ Qh, const bf16_t* __restrict__ Kh,
                 const bf16_t* __restrict__ VhT, bf16_t* __restrict__ Ctx) {
  __shared__ __align__(16) bf16_t Pl[2][32][32];  // double-buffered P staging
  const int lane = threadIdx.x;
  const int m = lane & 15, g = lane >> 4, n = m;
  const int qb = blockIdx.x;                      // 32-row query block
  const int bh = blockIdx.y;
  const int b = bh / H_, h = bh % H_;

  v16bf aQ[2][2];
#pragma unroll
  for (int mt = 0; mt < 2; ++mt) {
    const bf16_t* qrow = Qh + ((size_t)bh * S_ + qb * 32 + mt * 16 + m) * DK_;
    aQ[mt][0] = a_tile_bf16(qrow, 0, g);
    aQ[mt][1] = a_tile_bf16(qrow, 32, g);
  }
  const bf16_t* kbase = Kh  + (size_t)bh * S_ * DK_;
  const bf16_t* vbase = VhT + (size_t)bh * DV_ * S_;

  float mrow[2][8], lrow[2][8];
  v8f acc[2][4];
#pragma unroll
  for (int mt = 0; mt < 2; ++mt)
#pragma unroll
    for (int r = 0; r < 8; ++r) { mrow[mt][r] = -1e30f; lrow[mt][r] = 0.0f; }
#pragma unroll
  for (int mt = 0; mt < 2; ++mt)
#pragma unroll
    for (int vt = 0; vt < 4; ++vt) acc[mt][vt] = (v8f){0};

  const int tEnd = qb * 32 + 32;                  // causal horizon (exclusive)
  for (int tb = 0; tb < tEnd; tb += 32) {
    const int pb = (tb >> 5) & 1;                 // LDS ping-pong buffer
    // ---- K-tile B-operands, shared by both M-tiles -----------------------
    v16bf bk[2][2];                               // [t-tile][k-step]
    bk[0][0] = *(const v16bf*)(kbase + (size_t)(tb + n) * DK_ + 16 * g);
    bk[0][1] = *(const v16bf*)(kbase + (size_t)(tb + n) * DK_ + 32 + 16 * g);
    bk[1][0] = *(const v16bf*)(kbase + (size_t)(tb + 16 + n) * DK_ + 16 * g);
    bk[1][1] = *(const v16bf*)(kbase + (size_t)(tb + 16 + n) * DK_ + 32 + 16 * g);

    // ---- prefetch next iteration's K/V while this one computes -----------
    if (tb + 32 < tEnd) {                         // wave-uniform condition
      __builtin_prefetch(kbase + (size_t)(tb + 32 + n) * DK_, 0, 1);
      __builtin_prefetch(kbase + (size_t)(tb + 48 + n) * DK_, 0, 1);
#pragma unroll
      for (int vt = 0; vt < 4; ++vt)
        __builtin_prefetch(vbase + (size_t)(vt * 16 + n) * S_ + tb + 32, 0, 1);
    }

#pragma unroll
    for (int mt = 0; mt < 2; ++mt) {
      v8f c0 = (v8f){0}, c1 = (v8f){0};
      c0 = wmma_bf16(aQ[mt][0], bk[0][0], c0);
      c0 = wmma_bf16(aQ[mt][1], bk[0][1], c0);
      c1 = wmma_bf16(aQ[mt][0], bk[1][0], c1);
      c1 = wmma_bf16(aQ[mt][1], bk[1][1], c1);
      // ---- causal mask + online softmax (log2 domain) --------------------
#pragma unroll
      for (int r = 0; r < 8; ++r) {
        const int srow = qb * 32 + mt * 16 + r + 8 * g;
        float v0 = (tb + n      <= srow) ? c0[r] : -1e9f;
        float v1 = (tb + 16 + n <= srow) ? c1[r] : -1e9f;
        float mx = fmaxf(v0, v1);
#pragma unroll
        for (int off = 8; off; off >>= 1) mx = fmaxf(mx, __shfl_xor(mx, off, 16));
        const float mnew = fmaxf(mrow[mt][r], mx);
        const float scl  = exp2f(mrow[mt][r] - mnew);
        mrow[mt][r] = mnew;
        const float p0 = exp2f(v0 - mnew);
        const float p1 = exp2f(v1 - mnew);
        float rs = p0 + p1;
#pragma unroll
        for (int off = 8; off; off >>= 1) rs += __shfl_xor(rs, off, 16);
        lrow[mt][r] = lrow[mt][r] * scl + rs;
#pragma unroll
        for (int vt = 0; vt < 4; ++vt) acc[mt][vt][r] = acc[mt][vt][r] * scl;
        Pl[pb][mt * 16 + r + 8 * g][n]      = (__bf16)p0;
        Pl[pb][mt * 16 + r + 8 * g][16 + n] = (__bf16)p1;
      }
    }
    __syncthreads();   // single-wave WG: lowers to s_wait_dscnt before reads
    // ---- P (32x32 bf16) @ V (32x64): V-tiles shared by both M-tiles ------
    const v16bf aP0 = a_tile_bf16(&Pl[pb][m][0], 0, g);
    const v16bf aP1 = a_tile_bf16(&Pl[pb][16 + m][0], 0, g);
#pragma unroll
    for (int vt = 0; vt < 4; ++vt) {
      v16bf bv = *(const v16bf*)(vbase + (size_t)(vt * 16 + n) * S_ + tb + 16 * g);
      acc[0][vt] = wmma_bf16(aP0, bv, acc[0][vt]);
      acc[1][vt] = wmma_bf16(aP1, bv, acc[1][vt]);
    }
    // no trailing barrier: next iteration writes the other Pl buffer
  }
  // ---- normalize and emit token-major bf16 context [B,S,H*DV] ------------------------
#pragma unroll
  for (int mt = 0; mt < 2; ++mt)
#pragma unroll
    for (int r = 0; r < 8; ++r) {
      const float inv = 1.0f / lrow[mt][r];
      const int srow = qb * 32 + mt * 16 + r + 8 * g;
      bf16_t* dst = Ctx + ((size_t)b * S_ + srow) * HD_ + h * DV_;
#pragma unroll
      for (int vt = 0; vt < 4; ++vt)
        dst[vt * 16 + n] = (__bf16)(acc[mt][vt][r] * inv);
    }
}

// ---------------------------------------------------------------------------
// Kernel 3: output projection  Ctx[B*S,1024] x WObf[1024,1024]^T -> fp32
// 64x64 tile per wave.
// ---------------------------------------------------------------------------
__global__ __launch_bounds__(32)
void oproj_kernel(const bf16_t* __restrict__ Ctx, const bf16_t* __restrict__ WObf,
                  float* __restrict__ Out) {
  const int lane = threadIdx.x;
  const int m = lane & 15, g = lane >> 4, n = m;
  const size_t rb = (size_t)blockIdx.x * 64;
  const int nb = blockIdx.y * 64;

  v8f acc[4][4];
#pragma unroll
  for (int mt = 0; mt < 4; ++mt)
#pragma unroll
    for (int nt = 0; nt < 4; ++nt) acc[mt][nt] = (v8f){0};

  for (int cb = 0; cb < HD_; cb += 32) {
    v16bf a[4];
#pragma unroll
    for (int mt = 0; mt < 4; ++mt)
      a[mt] = a_tile_bf16(Ctx + (rb + mt * 16 + m) * HD_, cb, g);
#pragma unroll
    for (int nt = 0; nt < 4; ++nt) {
      v16bf b = *(const v16bf*)(WObf + (size_t)(nb + nt * 16 + n) * HD_ + cb + 16 * g);
#pragma unroll
      for (int mt = 0; mt < 4; ++mt) acc[mt][nt] = wmma_bf16(a[mt], b, acc[mt][nt]);
    }
  }
#pragma unroll
  for (int mt = 0; mt < 4; ++mt)
#pragma unroll
    for (int nt = 0; nt < 4; ++nt)
#pragma unroll
      for (int r = 0; r < 8; ++r)
        Out[(rb + mt * 16 + r + 8 * g) * (size_t)D_ + nb + nt * 16 + n] =
            acc[mt][nt][r];
}

// ---------------------------------------------------------------------------
extern "C" void kernel_launch(void* const* d_in, const int* in_sizes, int n_in,
                              void* d_out, int out_size, void* d_ws, size_t ws_size,
                              hipStream_t stream) {
  (void)in_sizes; (void)n_in; (void)out_size; (void)ws_size;
  const float* Q  = (const float*)d_in[0];
  const float* K  = (const float*)d_in[1];
  const float* V  = (const float*)d_in[2];
  const float* WQ = (const float*)d_in[3];
  const float* WK = (const float*)d_in[4];
  const float* WV = (const float*)d_in[5];
  const float* WO = (const float*)d_in[6];
  float* Out = (float*)d_out;

  // workspace layout (bf16):
  //   4 x 1M-elem weights (8 MiB) + 4 x 8.4M-elem activations (64 MiB) = 72 MiB
  const size_t nW  = (size_t)1024 * 1024;          // H*DK*D == D*HD == 1M
  const size_t nPH = (size_t)B_ * H_ * S_ * DK_;   // 8.4M
  bf16_t* WQb = (bf16_t*)d_ws;
  bf16_t* WKb = WQb + nW;
  bf16_t* WVb = WKb + nW;
  bf16_t* WOb = WVb + nW;
  bf16_t* Qh  = WOb + nW;
  bf16_t* Kh  = Qh + nPH;
  bf16_t* VhT = Kh + nPH;
  bf16_t* Ctx = VhT + nPH;

  const dim3 blk32(32, 1, 1);
  const dim3 blk256(256, 1, 1);
  const dim3 gCvt(nW / (8 * 256), 1);              // 512 blocks
  const dim3 gGemm((B_ * S_) / 64, D_ / 64);       // 128 x 16
  // fold 1/sqrt(DK) and log2(e) into Qh so softmax uses exp2
  const float qscale = 0.125f * 1.4426950408889634f;

  hipLaunchKernelGGL(cvt_kernel, gCvt, blk256, 0, stream, WQ, WQb);
  hipLaunchKernelGGL(cvt_kernel, gCvt, blk256, 0, stream, WK, WKb);
  hipLaunchKernelGGL(cvt_kernel, gCvt, blk256, 0, stream, WV, WVb);
  hipLaunchKernelGGL(cvt_kernel, gCvt, blk256, 0, stream, WO, WOb);

  hipLaunchKernelGGL(proj_kernel, gGemm, blk32, 0, stream, Q, WQb, Qh, 0, qscale);
  hipLaunchKernelGGL(proj_kernel, gGemm, blk32, 0, stream, K, WKb, Kh, 0, 1.0f);
  hipLaunchKernelGGL(proj_kernel, gGemm, blk32, 0, stream, V, WVb, VhT, 1, 1.0f);

  hipLaunchKernelGGL(attn_kernel, dim3(S_ / 32, B_ * H_), blk32, 0, stream,
                     Qh, Kh, VhT, Ctx);
  hipLaunchKernelGGL(oproj_kernel, gGemm, blk32, 0, stream, Ctx, WOb, Out);
}
